// PackedLinear_63410897158504
// MI455X (gfx1250) — matrix operand order repaired and checked by reading, not compiled, not run on hardware
//
#include <hip/hip_runtime.h>

// ---------------------------------------------------------------------------
// PackedLinear on MI455X (gfx1250, wave32):
//   out = scatter(gather(x)[4096,4096] @ W^T + bias)  with fp32 data.
// Strategy: bf16x3 (Dekker split) WMMA GEMM with fp32 accumulation.
//   x = hi + lo exactly (hi = truncate-to-bf16), product approximated by
//   hi*hi + hi*lo + lo*hi  -> ~fp32 accuracy at 3x bf16 matrix rate, ~2.7x
//   faster than the native V_WMMA_F32_16X16X4_F32 (K=4) path.
// Block tile 128x128, K-step 32, 8 waves (2 M x 4 N), wave tile 64x32:
//   24 v_wmma_f32_16x16x32_bf16 per wave per K-step, double-buffered LDS.
// ---------------------------------------------------------------------------

namespace {
constexpr int kDin  = 4096;
constexpr int kDout = 4096;
constexpr int kNAct = 4096;
constexpr int kRows = 8192;          // B*S
constexpr int kBK   = 32;            // WMMA bf16 K
constexpr int kNK   = kDin / kBK;    // 128 K-steps
constexpr int kBM   = 128;           // block tile M (packed rows)
constexpr int kBN   = 128;           // block tile N (dout cols)
constexpr int kLdk  = kBK + 8;       // padded LDS row (40 bf16 = 80B, bank-conflict free)
}

typedef __attribute__((ext_vector_type(16))) __bf16 v16bf;
typedef __attribute__((ext_vector_type(8)))  float  v8f;

union FragU { v16bf v; uint4 q[2]; };

// Split 4 floats into packed bf16 hi (truncated, exact residual) and bf16 lo.
__device__ __forceinline__ void split4(const float4 v,
                                       unsigned long long& hq,
                                       unsigned long long& lq)
{
  const float f[4] = {v.x, v.y, v.z, v.w};
  unsigned long long h = 0ull, l = 0ull;
#pragma unroll
  for (int i = 0; i < 4; ++i) {
    const unsigned u  = __float_as_uint(f[i]);
    const unsigned uh = u & 0xffff0000u;            // truncate to bf16: residual exact
    const float    r  = f[i] - __uint_as_float(uh);
    const unsigned short hs = (unsigned short)(uh >> 16);
    const unsigned short ls = __builtin_bit_cast(unsigned short, (__bf16)r);
    h |= (unsigned long long)hs << (16 * i);
    l |= (unsigned long long)ls << (16 * i);
  }
  hq = h; lq = l;
}

__global__ __launch_bounds__(256) void zero_out_f4(float4* __restrict__ p, int n4)
{
  const int i = blockIdx.x * 256 + threadIdx.x;
  if (i < n4) p[i] = float4{0.f, 0.f, 0.f, 0.f};
}

__global__ __launch_bounds__(256) void packed_linear_wmma_bf16x3(
    const float*     __restrict__ x,            // [8192, 4096] f32
    const long long* __restrict__ active_rows,  // [4096] i64
    const float*     __restrict__ weight,       // [4096, 4096] f32 (DOUT, DIN)
    const float*     __restrict__ bias,         // [4096] f32
    float*           __restrict__ out)          // [8192, 4096] f32 (pre-zeroed)
{
  __shared__ __align__(16) __bf16 sAh[2][kBM][kLdk];
  __shared__ __align__(16) __bf16 sAl[2][kBM][kLdk];
  __shared__ __align__(16) __bf16 sBh[2][kBN][kLdk];
  __shared__ __align__(16) __bf16 sBl[2][kBN][kLdk];
  __shared__ int sRowIdx[kBM];

  const int t    = threadIdx.x;
  const int lane = t & 31;
  const int wv   = t >> 5;           // wave 0..7  (wave32)
  const int wm   = (wv & 1) * 64;    // wave M offset in block: 0,64
  const int wn   = (wv >> 1) * 32;   // wave N offset in block: 0,32,64,96
  const int hh   = lane >> 4;        // lane half (ISA 16-bit WMMA layouts)
  const int li   = lane & 15;

  const int m0 = blockIdx.y * kBM;
  const int n0 = blockIdx.x * kBN;

  if (t < kBM) sRowIdx[t] = (int)active_rows[m0 + t];
  __syncthreads();

  float4 ra[4];  // 128x32 f32 A tile: 1024 float4, 4 per thread
  float4 rb[4];  // 128x32 f32 B tile: 1024 float4, 4 per thread

  auto load_global = [&](int k0) {
#pragma unroll
    for (int i = 0; i < 4; ++i) {
      const int f = t + 256 * i;
      const int r = f >> 3, c = f & 7;
      ra[i] = *(const float4*)(x + (size_t)sRowIdx[r] * kDin + k0 + 4 * c);
    }
#pragma unroll
    for (int i = 0; i < 4; ++i) {
      const int f = t + 256 * i;
      const int r = f >> 3, c = f & 7;
      rb[i] = *(const float4*)(weight + (size_t)(n0 + r) * kDin + k0 + 4 * c);
    }
  };

  auto stage_lds = [&](int buf) {
#pragma unroll
    for (int i = 0; i < 4; ++i) {
      const int f = t + 256 * i;
      const int r = f >> 3, c = f & 7;
      unsigned long long hq, lq;
      split4(ra[i], hq, lq);
      *(unsigned long long*)&sAh[buf][r][c * 4] = hq;
      *(unsigned long long*)&sAl[buf][r][c * 4] = lq;
    }
#pragma unroll
    for (int i = 0; i < 4; ++i) {
      const int f = t + 256 * i;
      const int r = f >> 3, c = f & 7;
      unsigned long long hq, lq;
      split4(rb[i], hq, lq);
      *(unsigned long long*)&sBh[buf][r][c * 4] = hq;
      *(unsigned long long*)&sBl[buf][r][c * 4] = lq;
    }
  };

  v8f acc[4][2];
#pragma unroll
  for (int a = 0; a < 4; ++a)
#pragma unroll
    for (int b = 0; b < 2; ++b)
#pragma unroll
      for (int j = 0; j < 8; ++j) acc[a][b][j] = 0.0f;

  // software pipeline: global->regs, regs->(split)->LDS, double-buffered
  load_global(0);
  stage_lds(0);
  __syncthreads();

  for (int ks = 0; ks < kNK; ++ks) {
    const int buf = ks & 1;
    if (ks + 1 < kNK) load_global((ks + 1) * kBK);

    // A fragments (16x32 bf16): lane li = row M; half hh selects K chunks
    // {8h..8h+7} and {8h+16..8h+23}  (ISA 16-bit A layout)
    FragU fah[4], fal[4], fbh[2], fbl[2];
#pragma unroll
    for (int a = 0; a < 4; ++a) {
      const int r = wm + a * 16 + li;
      const __bf16* ph = &sAh[buf][r][0];
      const __bf16* pl = &sAl[buf][r][0];
      fah[a].q[0] = *(const uint4*)(ph + 8 * hh);
      fah[a].q[1] = *(const uint4*)(ph + 8 * hh + 16);
      fal[a].q[0] = *(const uint4*)(pl + 8 * hh);
      fal[a].q[1] = *(const uint4*)(pl + 8 * hh + 16);
    }
    // B fragments (32x16 bf16): lane li = col N; half hh selects K chunks
    // {8h..8h+7} and {8h+16..8h+23}  (same K order as A)
#pragma unroll
    for (int b = 0; b < 2; ++b) {
      const int r = wn + b * 16 + li;  // weight row = output column
      const __bf16* ph = &sBh[buf][r][8 * hh];
      const __bf16* pl = &sBl[buf][r][8 * hh];
      fbh[b].q[0] = *(const uint4*)(ph);
      fbh[b].q[1] = *(const uint4*)(ph + 16);
      fbl[b].q[0] = *(const uint4*)(pl);
      fbl[b].q[1] = *(const uint4*)(pl + 16);
    }

#pragma unroll
    for (int a = 0; a < 4; ++a)
#pragma unroll
      for (int b = 0; b < 2; ++b) {
        acc[a][b] = __builtin_amdgcn_wmma_f32_16x16x32_bf16(
            false, fah[a].v, false, fbh[b].v, (short)0, acc[a][b], false, false);
        acc[a][b] = __builtin_amdgcn_wmma_f32_16x16x32_bf16(
            false, fah[a].v, false, fbl[b].v, (short)0, acc[a][b], false, false);
        acc[a][b] = __builtin_amdgcn_wmma_f32_16x16x32_bf16(
            false, fal[a].v, false, fbh[b].v, (short)0, acc[a][b], false, false);
      }

    if (ks + 1 < kNK) stage_lds(buf ^ 1);
    __syncthreads();
  }

  // Epilogue: bias add + scatter to active global rows.
  // C/D layout: VGPR j -> M = j + 8*hh, N = li.
#pragma unroll
  for (int b = 0; b < 2; ++b) {
    const int ncol = n0 + wn + b * 16 + li;
    const float bv = bias[ncol];
#pragma unroll
    for (int a = 0; a < 4; ++a) {
#pragma unroll
      for (int j = 0; j < 8; ++j) {
        const int mrow = wm + a * 16 + j + 8 * hh;
        const int g    = sRowIdx[mrow];
        out[(size_t)g * kDout + ncol] = acc[a][b][j] + bv;
      }
    }
  }
}

extern "C" void kernel_launch(void* const* d_in, const int* in_sizes, int n_in,
                              void* d_out, int out_size, void* d_ws, size_t ws_size,
                              hipStream_t stream)
{
  (void)in_sizes; (void)n_in; (void)out_size; (void)d_ws; (void)ws_size;

  const float*     x    = (const float*)d_in[0];      // [4,2048,4096] f32
  const long long* rows = (const long long*)d_in[1];  // [4096] i64
  const float*     wgt  = (const float*)d_in[2];      // [4096,4096] f32
  const float*     bias = (const float*)d_in[3];      // [4096] f32
  float*           out  = (float*)d_out;              // [4,2048,4096] f32

  // 1) zero all output rows (inactive rows must be exactly 0, no bias)
  const int n4 = kRows * kDout / 4;                   // 8,388,608 float4s
  zero_out_f4<<<(n4 + 255) / 256, 256, 0, stream>>>((float4*)out, n4);

  // 2) gathered GEMM + bias + scatter
  dim3 grid(kDout / kBN, kNAct / kBM);                // (32, 32)
  packed_linear_wmma_bf16x3<<<grid, 256, 0, stream>>>(x, rows, wgt, bias, out);
}
